// Attention_27238682591400
// MI455X (gfx1250) — matrix operand order repaired and checked by reading, not compiled, not run on hardware
//
#include <hip/hip_runtime.h>
#include <hip/hip_bf16.h>
#include <stdint.h>

#define DI __device__ __forceinline__

// ---------- problem constants ----------
constexpr int cB = 4, cT0 = 1024, cT1 = 256, cT = 1280;
constexpr int cSC = 1024, cS = 2304;
constexpr int cW0 = 2048, cW1 = 1024;
constexpr int cN = 8, cH = 256;

// ---------- WMMA / TDM types ----------
typedef __attribute__((ext_vector_type(16))) __bf16 v16bf;
typedef __attribute__((ext_vector_type(8)))  float  v8f;
typedef __attribute__((ext_vector_type(4)))  unsigned v4u;
typedef __attribute__((ext_vector_type(8)))  int    v8i;
typedef __attribute__((ext_vector_type(4)))  int    v4i;

struct alignas(16) U4 { unsigned x, y, z, w; };
struct alignas(16) F16x2 { U4 a, b; };   // 32 bytes == v16bf

DI v16bf mkfrag(U4 a, U4 b) {
    F16x2 t; t.a = a; t.b = b;
    return __builtin_bit_cast(v16bf, t);
}

DI unsigned short f2bfu(float f) {              // round-to-nearest-even f32 -> bf16 bits
    union { float f; unsigned u; } v; v.f = f;
    unsigned r = v.u + 0x7FFFu + ((v.u >> 16) & 1u);
    return (unsigned short)(r >> 16);
}

DI unsigned lds_off(const void* p) {            // LDS byte offset (low 32 bits of generic addr)
    return (unsigned)(size_t)p;
}

DI void async_copy_b128(unsigned lds_dst, const void* gsrc) {
    unsigned long long ga = (unsigned long long)gsrc;
    asm volatile("global_load_async_to_lds_b128 %0, %1, off"
                 :: "v"(lds_dst), "v"(ga) : "memory");
}
DI void async_copy_b128_o16(unsigned lds_dst, const void* gsrc) {
    unsigned long long ga = (unsigned long long)gsrc;
    asm volatile("global_load_async_to_lds_b128 %0, %1, off offset:16"
                 :: "v"(lds_dst), "v"(ga) : "memory");
}
DI void wait_async0() { asm volatile("s_wait_asynccnt 0x0" ::: "memory"); }

// ---------- elementwise f32 -> bf16 ----------
__global__ void k_f32_to_bf16(const float* __restrict__ src,
                              unsigned short* __restrict__ dst, long long n) {
    long long i = (long long)blockIdx.x * blockDim.x + threadIdx.x;
    long long st = (long long)gridDim.x * blockDim.x;
    for (; i < n; i += st) dst[i] = f2bfu(src[i]);
}

// ---------- generic batched bf16 GEMM: C(f32) = A(MxK) * B(KxN) ----------
// grid = (M/128, N/64, batches); batch z -> bb = z/innerN (A,C), nn = z%innerN (B,C)
// Double-buffered: A tiles via global_load_async_to_lds_b128 (ASYNCcnt),
// B tiles via VGPR transpose into LDS; staging overlaps WMMA on current tile.
__global__ __launch_bounds__(256, 2)
void k_gemm_bf16(const unsigned short* __restrict__ A, long long sAb, int lda,
                 const unsigned short* __restrict__ Bw, long long sBn, int ldb,
                 float* __restrict__ C, long long sCb, long long sCn, int ldc,
                 int K, int innerN) {
    __shared__ unsigned short Asub[2][128][40];   // pitch 40 (80B, 16B-aligned rows)
    __shared__ unsigned short Bt[2][64][40];      // transposed B tile: [n][k]

    const int tid = threadIdx.x;
    const int w = tid >> 5, L = tid & 31;
    const int lr = L & 15, hilo = L >> 4;
    const int z = blockIdx.z;
    const int bb = z / innerN, nn = z % innerN;
    A  += (long long)bb * sAb;
    Bw += (long long)nn * sBn;
    C  += (long long)bb * sCb + (long long)nn * sCn;
    const int m0 = blockIdx.x * 128;
    const int n0 = blockIdx.y * 64;

    const int arow = tid >> 1, ahalf = tid & 1;   // A staging role (32B per thread)
    const int brow = tid >> 3, bch = tid & 7;     // B staging role (16B per thread)

    auto issueA = [&](int k0, int buf) {
        unsigned dst = lds_off(&Asub[buf][arow][ahalf * 16]);
        const unsigned short* src = A + (long long)(m0 + arow) * lda + k0 + ahalf * 16;
        async_copy_b128(dst, src);
        async_copy_b128_o16(dst, src);
    };
    auto copyB = [&](int k0, int buf) {
        const U4* g = (const U4*)(Bw + (long long)(k0 + brow) * ldb + n0 + bch * 8);
        U4 v = g[0];
        union { U4 v; unsigned short s[8]; } u; u.v = v;
        #pragma unroll
        for (int e = 0; e < 8; ++e) Bt[buf][bch * 8 + e][brow] = u.s[e];
    };

    v8f acc[4] = {};
    const int nk = K >> 5;

    issueA(0, 0);
    copyB(0, 0);

    for (int i = 0; i < nk; ++i) {
        const int cur = i & 1;
        wait_async0();                       // current A tile landed in LDS
        __syncthreads();                     // + B ds_stores visible, prev readers done
        if (i + 1 < nk) {
            issueA(32 * (i + 1), 1 - cur);   // overlap next-tile DMA with compute
            copyB(32 * (i + 1), 1 - cur);
            if (i + 2 < nk)                  // global_prefetch_b8
                __builtin_prefetch(Bw + (long long)(32 * (i + 2) + brow) * ldb + n0, 0, 1);
        }
        const int ar = 16 * w + lr, koff = 8 * hilo;
        U4 a0 = *(const U4*)&Asub[cur][ar][koff];
        U4 a1 = *(const U4*)&Asub[cur][ar][16 + koff];
        v16bf af = mkfrag(a0, a1);
        #pragma unroll
        for (int j = 0; j < 4; ++j) {
            const int bc = 16 * j + lr, kb = 8 * hilo;
            U4 b0 = *(const U4*)&Bt[cur][bc][kb];
            U4 b1 = *(const U4*)&Bt[cur][bc][kb + 16];
            v16bf bf = mkfrag(b0, b1);
            acc[j] = __builtin_amdgcn_wmma_f32_16x16x32_bf16(
                false, af, false, bf, (short)0, acc[j], false, false);
        }
    }

    #pragma unroll
    for (int j = 0; j < 4; ++j)
        #pragma unroll
        for (int r = 0; r < 8; ++r) {
            int row = m0 + 16 * w + r + 8 * hilo;
            int col = n0 + 16 * j + lr;
            C[(long long)row * ldc + col] = acc[j][r];
        }
}

// ---------- RoPE on q (scaled by H^-0.5), f32 -> bf16 ----------
__global__ void k_rope_q(const float* __restrict__ qraw, const float* __restrict__ pos,
                         unsigned short* __restrict__ qb) {
    const long long total = (long long)cB * cT * cN * 128;
    long long i = (long long)blockIdx.x * blockDim.x + threadIdx.x;
    long long st = (long long)gridDim.x * blockDim.x;
    for (; i < total; i += st) {
        int hp = (int)(i & 127);
        long long btn = i >> 7;              // b*T*N + t*N + n
        long long bt = btn / cN;             // b*T + t
        float p = pos[bt];
        float inv = __expf(-(float)hp * (9.2103403719761836f / 128.0f)); // 10000^(-h/128)
        float s, c; __sincosf(p * inv, &s, &c);
        const float* base = qraw + btn * cH;
        float x1 = base[hp], x2 = base[hp + 128];
        unsigned short* ob = qb + btn * cH;
        ob[hp]       = f2bfu((x1 * c - x2 * s) * 0.0625f);
        ob[hp + 128] = f2bfu((x2 * c + x1 * s) * 0.0625f);
    }
}

// ---------- RoPE on k_new: f32 -> d_out k region (f32) + bf16 cache ----------
__global__ void k_rope_k(const float* __restrict__ kraw, const float* __restrict__ pos,
                         float* __restrict__ kout, unsigned short* __restrict__ kb) {
    const long long total = (long long)cB * cT * 128;
    long long i = (long long)blockIdx.x * blockDim.x + threadIdx.x;
    long long st = (long long)gridDim.x * blockDim.x;
    for (; i < total; i += st) {
        int hp = (int)(i & 127);
        long long bt = i >> 7;
        long long b = bt / cT, t = bt % cT;
        float p = pos[bt];
        float inv = __expf(-(float)hp * (9.2103403719761836f / 128.0f));
        float s, c; __sincosf(p * inv, &s, &c);
        const float* base = kraw + bt * cH;
        float x1 = base[hp], x2 = base[hp + 128];
        float o1 = x1 * c - x2 * s, o2 = x2 * c + x1 * s;
        long long so = (b * cS + cSC + t) * cH;
        kout[so + hp] = o1;        kout[so + hp + 128] = o2;
        kb[so + hp] = f2bfu(o1);   kb[so + hp + 128] = f2bfu(o2);
    }
}

// ---------- copy cache (k or v) into output + bf16 working copy ----------
__global__ void k_cache_copy(const float* __restrict__ cache, float* __restrict__ outp,
                             unsigned short* __restrict__ wb) {
    const long long total = (long long)cB * cSC * cH;
    long long i = (long long)blockIdx.x * blockDim.x + threadIdx.x;
    long long st = (long long)gridDim.x * blockDim.x;
    for (; i < total; i += st) {
        long long b = i / ((long long)cSC * cH);
        long long rem = i - b * ((long long)cSC * cH);
        long long dst = b * ((long long)cS * cH) + rem;
        float v = cache[i];
        outp[dst] = v; wb[dst] = f2bfu(v);
    }
}

// ---------- v_new passthrough: f32 -> d_out v region + bf16 ----------
__global__ void k_v_new(const float* __restrict__ vraw, float* __restrict__ vout,
                        unsigned short* __restrict__ vb) {
    const long long total = (long long)cB * cT * cH;
    long long i = (long long)blockIdx.x * blockDim.x + threadIdx.x;
    long long st = (long long)gridDim.x * blockDim.x;
    for (; i < total; i += st) {
        int h = (int)(i & 255);
        long long bt = i >> 8;
        long long b = bt / cT, t = bt % cT;
        long long dst = (b * cS + cSC + t) * cH + h;
        float v = vraw[i];
        vout[dst] = v; vb[dst] = f2bfu(v);
    }
}

// ---------- flash attention: grid (T/64, N, B), block 128 (4 waves) ----------
// K tile staged by the Tensor Data Mover (tensor_load_to_lds, TENSORcnt);
// the D# pad fields produce the padded 264-element LDS row pitch directly.
__global__ __launch_bounds__(128, 1)
void k_attn(const unsigned short* __restrict__ qb,
            const unsigned short* __restrict__ kb,
            const unsigned short* __restrict__ vb,
            unsigned short* __restrict__ enc) {
    __shared__ unsigned short Klds[32][264];     // K tile, row-major [s][h], TDM-filled
    __shared__ unsigned short Vt[256][40];       // V tile transposed [h][s]
    __shared__ unsigned short Plds[4][16][40];   // per-wave P staging

    const int tid = threadIdx.x;
    const int w = tid >> 5, L = tid & 31;
    const int lr = L & 15, hilo = L >> 4;
    const int t0 = blockIdx.x * 64;
    const int n = blockIdx.y;
    const int b = blockIdx.z;

    // Q fragments: wave w owns rows [t0+16w, t0+16w+16)
    v16bf qf[8];
    {
        const long long qrow = ((long long)(b * cT + t0 + 16 * w + lr) * cN + n) * cH;
        const int koff = 8 * hilo;
        #pragma unroll
        for (int kt = 0; kt < 8; ++kt) {
            U4 a0 = *(const U4*)(qb + qrow + kt * 32 + koff);
            U4 a1 = *(const U4*)(qb + qrow + kt * 32 + 16 + koff);
            qf[kt] = mkfrag(a0, a1);
        }
    }

    const unsigned short* Kg = kb + (long long)b * cS * cH;
    const unsigned short* Vg = vb + (long long)b * cS * cH;

    // D# group1 (constant): workgroup_mask=0, data_size=2B(code 1),
    // pad_enable, pad_interval=6 (512B = one 256-elem row), pad_amount=3 (16B),
    // tensor_dim0=256, tensor_dim1=2304, tile_dim0=256, tile_dim1=32,
    // tensor_dim0_stride=256.
    const v8i g1 = {0x07910000, 0x01000000, 0x09000000, 0x01000000, 32, 256, 0, 0};
    const v4i gz = {0, 0, 0, 0};
    const v8i gz8 = {0, 0, 0, 0, 0, 0, 0, 0};

    v8f o[16] = {};
    float m[8], l[8];
    #pragma unroll
    for (int r = 0; r < 8; ++r) { m[r] = -3.0e38f; l[r] = 0.0f; }

    const int s_end = cSC + t0 + 64;   // causal bound for this tile; multiple of 32, <= S

    for (int s0 = 0; s0 < s_end; s0 += 32) {
        __syncthreads();
        if (tid == 0) {                          // TDM: K tile (32 x 256) -> LDS
            unsigned long long ga = (unsigned long long)(size_t)(Kg + (long long)s0 * cH);
            v4u g0;
            g0[0] = 1u;                          // count=1, user mode
            g0[1] = lds_off(&Klds[0][0]);        // lds_addr
            g0[2] = (unsigned)ga;                // global_addr[31:0]
            g0[3] = (unsigned)(ga >> 32) | 0x80000000u;  // addr[56:32] | type=2
            __builtin_amdgcn_tensor_load_to_lds(g0, g1, gz, gz, gz8, 0);
        }
        #pragma unroll
        for (int it = 0; it < 8; ++it) {         // V tile 32x256 -> transposed LDS
            int lin = tid + 128 * it;            // 0..1023 u4-chunks
            int row = lin >> 5, c4 = lin & 31;
            U4 vv = *(const U4*)(Vg + (long long)(s0 + row) * cH + c4 * 8);
            union { U4 v; unsigned short s[8]; } u; u.v = vv;
            #pragma unroll
            for (int e = 0; e < 8; ++e) Vt[c4 * 8 + e][row] = u.s[e];
        }
        if (s0 + 32 < s_end)
            __builtin_prefetch(Vg + (long long)(s0 + 32 + (tid >> 2)) * cH, 0, 1);
        if (tid == 0) __builtin_amdgcn_s_wait_tensorcnt(0);
        __syncthreads();

        // logits = Q (16xH) x K^T : two 16x16 S-tiles
        v8f cj[2] = {};
        #pragma unroll
        for (int j = 0; j < 2; ++j) {
            #pragma unroll
            for (int kt = 0; kt < 8; ++kt) {
                int srow = 16 * j + lr;
                int hb = kt * 32 + 8 * hilo;
                U4 b0 = *(const U4*)&Klds[srow][hb];
                U4 b1 = *(const U4*)&Klds[srow][hb + 16];
                v16bf bf = mkfrag(b0, b1);
                cj[j] = __builtin_amdgcn_wmma_f32_16x16x32_bf16(
                    false, qf[kt], false, bf, (short)0, cj[j], false, false);
            }
        }

        // causal mask + online softmax (row reductions across 16 lanes)
        #pragma unroll
        for (int r = 0; r < 8; ++r) {
            int trow = t0 + 16 * w + r + 8 * hilo;
            int qp = cSC + trow;
            float v0 = (s0 + lr      <= qp) ? cj[0][r] : -1.0e30f;
            float v1 = (s0 + 16 + lr <= qp) ? cj[1][r] : -1.0e30f;
            float mx = fmaxf(v0, v1);
            #pragma unroll
            for (int off = 8; off > 0; off >>= 1)
                mx = fmaxf(mx, __shfl_xor(mx, off, 16));
            float mnew = fmaxf(m[r], mx);
            float alpha = __expf(m[r] - mnew);
            float p0 = __expf(v0 - mnew);
            float p1 = __expf(v1 - mnew);
            float rs = p0 + p1;
            #pragma unroll
            for (int off = 8; off > 0; off >>= 1)
                rs += __shfl_xor(rs, off, 16);
            l[r] = l[r] * alpha + rs;
            m[r] = mnew;
            #pragma unroll
            for (int ht = 0; ht < 16; ++ht) o[ht][r] *= alpha;
            Plds[w][r + 8 * hilo][lr]      = f2bfu(p0);
            Plds[w][r + 8 * hilo][16 + lr] = f2bfu(p1);
        }
        __syncthreads();

        // O += P (16x32) x V (32xH)
        {
            const int koff = 8 * hilo;
            U4 a0 = *(const U4*)&Plds[w][lr][koff];
            U4 a1 = *(const U4*)&Plds[w][lr][16 + koff];
            v16bf pf = mkfrag(a0, a1);
            #pragma unroll
            for (int ht = 0; ht < 16; ++ht) {
                int h = ht * 16 + lr;
                int kb16 = 8 * hilo;
                U4 b0 = *(const U4*)&Vt[h][kb16];
                U4 b1 = *(const U4*)&Vt[h][kb16 + 16];
                v16bf vf = mkfrag(b0, b1);
                o[ht] = __builtin_amdgcn_wmma_f32_16x16x32_bf16(
                    false, pf, false, vf, (short)0, o[ht], false, false);
            }
        }
    }

    // normalize + write encoded (B, T, N*H) bf16
    #pragma unroll
    for (int ht = 0; ht < 16; ++ht)
        #pragma unroll
        for (int r = 0; r < 8; ++r) {
            int trow = t0 + 16 * w + r + 8 * hilo;
            float val = o[ht][r] / l[r];
            enc[((long long)(b * cT + trow)) * (cN * cH) + n * cH + ht * 16 + lr] = f2bfu(val);
        }
}

// =====================================================================
extern "C" void kernel_launch(void* const* d_in, const int* in_sizes, int n_in,
                              void* d_out, int out_size, void* d_ws, size_t ws_size,
                              hipStream_t stream) {
    (void)in_sizes; (void)n_in; (void)out_size; (void)ws_size;
    const float* x0      = (const float*)d_in[0];
    const float* x1      = (const float*)d_in[1];
    const float* wq0     = (const float*)d_in[2];
    const float* wkv0    = (const float*)d_in[3];
    const float* wo0     = (const float*)d_in[4];
    const float* wq1     = (const float*)d_in[5];
    const float* wkv1    = (const float*)d_in[6];
    const float* wo1     = (const float*)d_in[7];
    const float* cache_k = (const float*)d_in[8];
    const float* cache_v = (const float*)d_in[9];
    const float* pos     = (const float*)d_in[10];

    float* out  = (float*)d_out;
    float* out0 = out;                                        // B*T0*W0 = 8388608
    float* out1 = out + (long long)cB * cT0 * cW0;            // + 1048576
    float* kout = out1 + (long long)cB * cT1 * cW1;           // B*S*H
    float* vout = kout + (long long)cB * cS * cH;

    char* ws = (char*)d_ws;
    size_t off = 0;
    auto alloc = [&](size_t bytes) -> void* {
        void* p = (void*)(ws + off);
        off += (bytes + 255) & ~(size_t)255;
        return p;
    };
    const long long nx0 = (long long)cB * cT0 * cW0, nx1 = (long long)cB * cT1 * cW1;
    const long long nwq0 = (long long)cN * cW0 * cH, nwkv0 = 2LL * cW0 * cH;
    const long long nwo0 = (long long)cN * cH * cW0;
    const long long nwq1 = (long long)cN * cW1 * cH, nwkv1 = 2LL * cW1 * cH;
    const long long nwo1 = (long long)cN * cH * cW1;
    const long long nq = (long long)cB * cT * cN * cH;
    const long long nk = (long long)cB * cT * cH;
    const long long nsv = (long long)cB * cS * cH;

    unsigned short* x0b   = (unsigned short*)alloc(nx0 * 2);
    unsigned short* x1b   = (unsigned short*)alloc(nx1 * 2);
    unsigned short* wq0b  = (unsigned short*)alloc(nwq0 * 2);
    unsigned short* wkv0b = (unsigned short*)alloc(nwkv0 * 2);
    unsigned short* wo0b  = (unsigned short*)alloc(nwo0 * 2);
    unsigned short* wq1b  = (unsigned short*)alloc(nwq1 * 2);
    unsigned short* wkv1b = (unsigned short*)alloc(nwkv1 * 2);
    unsigned short* wo1b  = (unsigned short*)alloc(nwo1 * 2);
    float*          q_raw = (float*)alloc(nq * 4);
    float*          k_raw = (float*)alloc(nk * 4);
    float*          v_raw = (float*)alloc(nk * 4);
    unsigned short* qb    = (unsigned short*)alloc(nq * 2);
    unsigned short* kbw   = (unsigned short*)alloc(nsv * 2);
    unsigned short* vbw   = (unsigned short*)alloc(nsv * 2);
    unsigned short* encb  = (unsigned short*)alloc(nq * 2);

    auto conv = [&](const float* s, unsigned short* d, long long n) {
        int blocks = (int)((n + 255) / 256); if (blocks > 16384) blocks = 16384;
        k_f32_to_bf16<<<blocks, 256, 0, stream>>>(s, d, n);
    };
    conv(x0, x0b, nx0);     conv(x1, x1b, nx1);
    conv(wq0, wq0b, nwq0);  conv(wkv0, wkv0b, nwkv0); conv(wo0, wo0b, nwo0);
    conv(wq1, wq1b, nwq1);  conv(wkv1, wkv1b, nwkv1); conv(wo1, wo1b, nwo1);

    // ---- projections (all WMMA GEMMs) ----
    k_gemm_bf16<<<dim3(8, 4, 32), 256, 0, stream>>>(
        x0b, (long long)cT0 * cW0, cW0, wq0b, (long long)cW0 * cH, cH,
        q_raw, (long long)cT * cN * cH, cH, cN * cH, cW0, cN);
    k_gemm_bf16<<<dim3(2, 4, 32), 256, 0, stream>>>(
        x1b, (long long)cT1 * cW1, cW1, wq1b, (long long)cW1 * cH, cH,
        q_raw + (long long)cT0 * cN * cH, (long long)cT * cN * cH, cH, cN * cH, cW1, cN);
    k_gemm_bf16<<<dim3(8, 4, 4), 256, 0, stream>>>(
        x0b, (long long)cT0 * cW0, cW0, wkv0b, 0, cH,
        k_raw, (long long)cT * cH, 0, cH, cW0, 1);
    k_gemm_bf16<<<dim3(8, 4, 4), 256, 0, stream>>>(
        x0b, (long long)cT0 * cW0, cW0, wkv0b + (long long)cW0 * cH, 0, cH,
        v_raw, (long long)cT * cH, 0, cH, cW0, 1);
    k_gemm_bf16<<<dim3(2, 4, 4), 256, 0, stream>>>(
        x1b, (long long)cT1 * cW1, cW1, wkv1b, 0, cH,
        k_raw + (long long)cT0 * cH, (long long)cT * cH, 0, cH, cW1, 1);
    k_gemm_bf16<<<dim3(2, 4, 4), 256, 0, stream>>>(
        x1b, (long long)cT1 * cW1, cW1, wkv1b + (long long)cW1 * cH, 0, cH,
        v_raw + (long long)cT0 * cH, (long long)cT * cH, 0, cH, cW1, 1);

    // ---- RoPE + cache assembly ----
    k_rope_q<<<20480, 256, 0, stream>>>(q_raw, pos, qb);
    k_rope_k<<<2560, 256, 0, stream>>>(k_raw, pos, kout, kbw);
    k_cache_copy<<<4096, 256, 0, stream>>>(cache_k, kout, kbw);
    k_cache_copy<<<4096, 256, 0, stream>>>(cache_v, vout, vbw);
    k_v_new<<<5120, 256, 0, stream>>>(v_raw, vout, vbw);

    // ---- flash attention (TDM-staged K tiles) ----
    k_attn<<<dim3(cT / 64, cN, cB), 128, 0, stream>>>(qb, kbw, vbw, encb);

    // ---- output projections ----
    k_gemm_bf16<<<dim3(8, 32, 4), 256, 0, stream>>>(
        encb, (long long)cT * cN * cH, cN * cH, wo0b, 0, cW0,
        out0, (long long)cT0 * cW0, 0, cW0, cN * cH, 1);
    k_gemm_bf16<<<dim3(2, 16, 4), 256, 0, stream>>>(
        encb + (long long)cT0 * cN * cH, (long long)cT * cN * cH, cN * cH, wo1b, 0, cW1,
        out1, (long long)cT1 * cW1, 0, cW1, cN * cH, 1);
}
